// MultiheadSlidingWindowSelfAttention_6356551598441
// MI455X (gfx1250) — compile-verified, matrix operand order checked by reading, not run on hardware
//
#include <hip/hip_runtime.h>
#include <hip/hip_bf16.h>

// ---------------------------------------------------------------------------
// Multihead sliding-window self-attention, CDNA5 (gfx1250), wave32 + WMMA f16.
//   T = 1024, E = 1024, H = 16, D = 64, band = +/-64 (129 wide)
// Pipeline:
//   1) f32 -> f16 conversion of activations + weights into workspace
//   2) WMMA GEMM: q,k,v = X @ W{q,k,v}^T + b   (f16 outputs)
//   3) V transposed + zero-padded to [h*D][T+192] for vectorized P@V operands
//   4) banded attention per (head, 16-query tile), WMMA scores + WMMA P@V
//   5) WMMA GEMM: out = ctx @ Wo^T + bo        (f32 output)
// ---------------------------------------------------------------------------

typedef _Float16 f16;
typedef __attribute__((ext_vector_type(16))) _Float16 v16h;
typedef __attribute__((ext_vector_type(8)))  float    v8f;

union Frag {
    v16h v;
    uint4 q[2];
};

// CDNA5 16-bit A/B operand layout (ISA 7.12.2):
//   lane<16 : row/col = lane,     elems 0..7 = K base+0..7, elems 8..15 = K base+16..23, base=0
//   lane>=16: row/col = lane-16,  same pattern with base=8
__device__ __forceinline__ v16h load_frag_rowmajor(const f16* base, int stride, int lane) {
    int r  = lane & 15;
    int kb = (lane >> 4) << 3;          // 0 or 8
    const f16* p = base + r * stride + kb;
    Frag f;
    f.q[0] = *reinterpret_cast<const uint4*>(p);
    f.q[1] = *reinterpret_cast<const uint4*>(p + 16);
    return f.v;
}

__device__ __forceinline__ v8f wmma_f16(v16h a, v16h b, v8f c) {
    return __builtin_amdgcn_wmma_f32_16x16x32_f16(
        /*neg_a=*/false, a, /*neg_b=*/false, b,
        /*c_mod=*/(short)0, c, /*reuse_a=*/false, /*reuse_b=*/false);
}

// ---------------------------------------------------------------------------
// f32 -> f16 conversion
// ---------------------------------------------------------------------------
__global__ void cvt_f32_to_f16(const float* __restrict__ src, f16* __restrict__ dst, int n) {
    int i = blockIdx.x * blockDim.x + threadIdx.x;
    if (i < n) dst[i] = (f16)src[i];
}

// ---------------------------------------------------------------------------
// V transpose + zero-pad:  Vt[hd][tp] = V[tp-PAD][hd]  (0 outside [0,T))
// hd = h*D + d == column e of the plain [t][e] layout. TS = T + 2*PAD.
// ---------------------------------------------------------------------------
__global__ void transpose_pad_v(const f16* __restrict__ Vh, f16* __restrict__ Vt, int nTotal) {
    constexpr int T = 1024, E = 1024, PAD = 96, TS = T + 2 * PAD;   // 1216
    int i = blockIdx.x * blockDim.x + threadIdx.x;
    if (i >= nTotal) return;
    int tp = i % TS;
    int hd = i / TS;
    int t  = tp - PAD;
    Vt[i] = (t >= 0 && t < T) ? Vh[(size_t)t * E + hd] : (f16)0.f;
}

// ---------------------------------------------------------------------------
// C[M,N] = A[M,K] @ W[N,K]^T + bias[N]
// Block tile 128x64, BK=32, 8 waves, each wave computes a 32x32 sub-tile
// (2x2 WMMA 16x16 accumulators). 128 v_wmma per wave over the K loop.
// ---------------------------------------------------------------------------
template <bool OUT_F16>
__global__ void __launch_bounds__(256)
gemm_xwt(const f16* __restrict__ A, const f16* __restrict__ W,
         const float* __restrict__ bias, void* __restrict__ Cout,
         int M, int N, int K) {
    constexpr int BM = 128, BN = 64, BK = 32;
    constexpr int AST = BK + 8;   // padded f16 stride (80B rows, 16B aligned)
    constexpr int BST = BK + 8;

    __shared__ __align__(16) f16 As[BM * AST];
    __shared__ __align__(16) f16 Bs[BN * BST];

    const int tid  = threadIdx.x;
    const int lane = tid & 31;
    const int wave = tid >> 5;          // 0..7
    const int wm   = wave >> 1;         // 0..3 (M)
    const int wn   = wave & 1;          // 0..1 (N)
    const int m0   = blockIdx.x * BM;
    const int n0   = blockIdx.y * BN;

    v8f acc[2][2] = {};

    for (int kk = 0; kk < K; kk += BK) {
        // Stage A tile: 128 rows x 32 k = 512 16B vectors, 2 per thread.
#pragma unroll
        for (int i = 0; i < 2; ++i) {
            int vid = i * 256 + tid;            // 0..511
            int r   = vid >> 2;
            int c8  = (vid & 3) * 8;
            *reinterpret_cast<uint4*>(&As[r * AST + c8]) =
                *reinterpret_cast<const uint4*>(&A[(size_t)(m0 + r) * K + kk + c8]);
        }
        // Stage W tile: 64 rows x 32 k = 256 16B vectors, 1 per thread.
        {
            int r  = tid >> 2;
            int c8 = (tid & 3) * 8;
            *reinterpret_cast<uint4*>(&Bs[r * BST + c8]) =
                *reinterpret_cast<const uint4*>(&W[(size_t)(n0 + r) * K + kk + c8]);
        }
        if (kk + BK < K) {
            __builtin_prefetch(&A[(size_t)(m0 + (tid & 127)) * K + kk + BK], 0, 1);
        }
        __syncthreads();

        v16h af[2], bf[2];
#pragma unroll
        for (int mi = 0; mi < 2; ++mi)
            af[mi] = load_frag_rowmajor(&As[(wm * 32 + mi * 16) * AST], AST, lane);
#pragma unroll
        for (int ni = 0; ni < 2; ++ni)
            bf[ni] = load_frag_rowmajor(&Bs[(wn * 32 + ni * 16) * BST], BST, lane);

#pragma unroll
        for (int mi = 0; mi < 2; ++mi)
#pragma unroll
            for (int ni = 0; ni < 2; ++ni)
                acc[mi][ni] = wmma_f16(af[mi], bf[ni], acc[mi][ni]);

        __syncthreads();
    }

    // Epilogue: C layout — VGPR i: lanes 0..15 -> M=i, lanes 16..31 -> M=i+8.
    const int mlo = (lane >> 4) << 3;
    const int nc  = lane & 15;
#pragma unroll
    for (int mi = 0; mi < 2; ++mi) {
#pragma unroll
        for (int ni = 0; ni < 2; ++ni) {
            int n = n0 + wn * 32 + ni * 16 + nc;
            float bn = bias[n];
#pragma unroll
            for (int i = 0; i < 8; ++i) {
                int m = m0 + wm * 32 + mi * 16 + mlo + i;
                float val = acc[mi][ni][i] + bn;
                if (OUT_F16)
                    reinterpret_cast<f16*>(Cout)[(size_t)m * N + n] = (f16)val;
                else
                    reinterpret_cast<float*>(Cout)[(size_t)m * N + n] = val;
            }
        }
    }
}

// ---------------------------------------------------------------------------
// Banded attention. One wave per (head, 16-query tile); 4 waves per block.
// Band for queries [t0, t0+15]: keys [t0-64, t0+79] -> 10 aligned 16-key tiles.
// ---------------------------------------------------------------------------
__global__ void __launch_bounds__(128)
attn_band(const f16* __restrict__ Q, const f16* __restrict__ Kd,
          const f16* __restrict__ Vt, f16* __restrict__ Ctx) {
    constexpr int T = 1024, E = 1024, D = 64;
    constexpr int PAD = 96, TS = T + 2 * PAD;  // padded V stride
    constexpr int NT = 10, BANDW = NT * 16;    // 160

    __shared__ __align__(16) float sc[4][16][BANDW];  // 40 KB
    __shared__ __align__(16) f16   pb[4][16][BANDW];  // 20 KB

    const int lane = threadIdx.x & 31;
    const int w    = threadIdx.x >> 5;
    const int task = blockIdx.x * 4 + w;      // 0..1023
    const int h    = task >> 6;
    const int t0   = (task & 63) << 4;
    const int sa   = t0 - 64;                 // aligned band start (can be <0)

    // Q fragments: 16 queries x D=64 -> two K=32 chunks.
    v16h qf[2];
    {
        int r  = lane & 15;
        int kb = (lane >> 4) << 3;
        const f16* qrow = Q + (size_t)(t0 + r) * E + h * D + kb;
        Frag f0, f1;
        f0.q[0] = *reinterpret_cast<const uint4*>(qrow);
        f0.q[1] = *reinterpret_cast<const uint4*>(qrow + 16);
        f1.q[0] = *reinterpret_cast<const uint4*>(qrow + 32);
        f1.q[1] = *reinterpret_cast<const uint4*>(qrow + 48);
        qf[0] = f0.v;
        qf[1] = f1.v;
    }

    // ---- scores = (Q @ K^T)/8, band-masked, into LDS ----
    for (int tile = 0; tile < NT; ++tile) {
        int s0 = sa + tile * 16;
        // B operand: 64(K=d) x 16(N=key). Per-lane column = one key row of Kd.
        int scol = s0 + (lane & 15);
        int scl  = scol < 0 ? 0 : (scol > T - 1 ? T - 1 : scol);
        int kb   = (lane >> 4) << 3;
        const f16* krow = Kd + (size_t)scl * E + h * D + kb;
        Frag b0, b1;
        b0.q[0] = *reinterpret_cast<const uint4*>(krow);
        b0.q[1] = *reinterpret_cast<const uint4*>(krow + 16);
        b1.q[0] = *reinterpret_cast<const uint4*>(krow + 32);
        b1.q[1] = *reinterpret_cast<const uint4*>(krow + 48);

        v8f a = {};
        a = wmma_f16(qf[0], b0.v, a);
        a = wmma_f16(qf[1], b1.v, a);

        int mlo = (lane >> 4) << 3;
        int n   = lane & 15;
        int s   = s0 + n;
#pragma unroll
        for (int i = 0; i < 8; ++i) {
            int m = mlo + i;
            int t = t0 + m;
            bool valid = (s >= 0) && (s < T) && (s >= t - 64) && (s <= t + 64);
            sc[w][m][tile * 16 + n] = valid ? a[i] * 0.125f : -__builtin_inff();
        }
    }
    __syncthreads();

    // ---- row softmax over the 160-wide band, all 32 lanes active ----
    {
        constexpr int HALFW = BANDW / 2;          // 80
        int m    = lane & 15;
        int half = lane >> 4;
        const float* row = &sc[w][m][half * HALFW];

        float mx = -__builtin_inff();
        for (int j = 0; j < HALFW; ++j) mx = fmaxf(mx, row[j]);
        mx = fmaxf(mx, __shfl_xor(mx, 16, 32));   // combine halves

        float sum = 0.f;
        for (int j = 0; j < HALFW; ++j) sum += __expf(row[j] - mx);
        sum += __shfl_xor(sum, 16, 32);

        float inv = 1.f / sum;
        f16* prow = &pb[w][m][half * HALFW];
        for (int j = 0; j < HALFW; ++j)
            prow[j] = (f16)(__expf(row[j] - mx) * inv);
    }
    __syncthreads();

    // ---- ctx = P(16x160) @ Vband(160x64): 5 K-chunks x 4 N-tiles ----
    // B operand comes from transposed+padded V: column d -> contiguous keys.
    v8f acc[4] = {};
    for (int c = 0; c < 5; ++c) {
        v16h pa;
        {
            int r  = lane & 15;
            int kb = (lane >> 4) << 3;
            const f16* pr = &pb[w][r][c * 32 + kb];
            Frag f;
            f.q[0] = *reinterpret_cast<const uint4*>(pr);
            f.q[1] = *reinterpret_cast<const uint4*>(pr + 16);
            pa = f.v;
        }
        int kb = (lane >> 4) << 3;
        int jb = c * 32 + kb;                      // key-chunk base for this lane
#pragma unroll
        for (int nt = 0; nt < 4; ++nt) {
            int d = nt * 16 + (lane & 15);
            const f16* vp = Vt + (size_t)(h * D + d) * TS + (sa + PAD + jb);
            Frag f;
            f.q[0] = *reinterpret_cast<const uint4*>(vp);        // keys jb..jb+7
            f.q[1] = *reinterpret_cast<const uint4*>(vp + 16);   // keys jb+16..jb+23
            acc[nt] = wmma_f16(pa, f.v, acc[nt]);
        }
    }

    // ---- write ctx (f16), plain [t][h*D+d] layout for the final GEMM ----
    const int mlo = (lane >> 4) << 3;
    const int nc  = lane & 15;
#pragma unroll
    for (int nt = 0; nt < 4; ++nt) {
#pragma unroll
        for (int i = 0; i < 8; ++i) {
            int m = mlo + i;
            Ctx[(size_t)(t0 + m) * E + h * D + nt * 16 + nc] = (f16)acc[nt][i];
        }
    }
}

// ---------------------------------------------------------------------------
// Launch
// ---------------------------------------------------------------------------
extern "C" void kernel_launch(void* const* d_in, const int* in_sizes, int n_in,
                              void* d_out, int out_size, void* d_ws, size_t ws_size,
                              hipStream_t stream) {
    const float* hs = (const float*)d_in[0];
    const float* Wq = (const float*)d_in[1];
    const float* bq = (const float*)d_in[2];
    const float* Wk = (const float*)d_in[3];
    const float* bk = (const float*)d_in[4];
    const float* Wv = (const float*)d_in[5];
    const float* bv = (const float*)d_in[6];
    const float* Wo = (const float*)d_in[7];
    const float* bo = (const float*)d_in[8];

    const int T = 1024, E = 1024;
    const int NE = T * E;                 // 1M elements per matrix
    const int TS = 1024 + 192;            // padded V stride
    const int NVT = E * TS;               // 1,245,184 f16 for Vt

    f16* ws  = (f16*)d_ws;
    f16* Xh  = ws + (size_t)0 * NE;
    f16* Wqh = ws + (size_t)1 * NE;
    f16* Wkh = ws + (size_t)2 * NE;
    f16* Wvh = ws + (size_t)3 * NE;
    f16* Woh = ws + (size_t)4 * NE;
    f16* qh  = ws + (size_t)5 * NE;
    f16* kh  = ws + (size_t)6 * NE;
    f16* vh  = ws + (size_t)7 * NE;
    f16* ch  = ws + (size_t)8 * NE;
    f16* vt  = ws + (size_t)9 * NE;       // ~20.5 MB total

    dim3 cb(256), cg(NE / 256);
    cvt_f32_to_f16<<<cg, cb, 0, stream>>>(hs, Xh, NE);
    cvt_f32_to_f16<<<cg, cb, 0, stream>>>(Wq, Wqh, NE);
    cvt_f32_to_f16<<<cg, cb, 0, stream>>>(Wk, Wkh, NE);
    cvt_f32_to_f16<<<cg, cb, 0, stream>>>(Wv, Wvh, NE);
    cvt_f32_to_f16<<<cg, cb, 0, stream>>>(Wo, Woh, NE);

    dim3 gb(256), gg(T / 128, E / 64);
    gemm_xwt<true><<<gg, gb, 0, stream>>>(Xh, Wqh, bq, qh, T, E, E);
    gemm_xwt<true><<<gg, gb, 0, stream>>>(Xh, Wkh, bk, kh, T, E, E);
    gemm_xwt<true><<<gg, gb, 0, stream>>>(Xh, Wvh, bv, vh, T, E, E);

    transpose_pad_v<<<dim3((NVT + 255) / 256), dim3(256), 0, stream>>>(vh, vt, NVT);

    attn_band<<<dim3(256), dim3(128), 0, stream>>>(qh, kh, vt, ch);

    gemm_xwt<false><<<gg, gb, 0, stream>>>(ch, Woh, bo, d_out, T, E, E);
}
